// OptimizedMoE_32658931319291
// MI455X (gfx1250) — matrix operand order repaired and checked by reading, not compiled, not run on hardware
//
#include <hip/hip_runtime.h>
#include <math.h>

// ---------------- Problem constants ----------------
#define N_TOK 8192
#define DDIM  1024
#define NEXP  8
#define HDIM  2048
#define ODIM  1024
#define NSLOT (N_TOK * 2)

// GEMM tiling
#define TM  128      // rows (slots) per block tile
#define TN  128      // cols per block tile
#define KC  32       // K chunk (bf16 WMMA K)
#define LDT 40       // padded LDS row stride in elements (80B -> bank-conflict free)

typedef unsigned short u16;
typedef __bf16 bf16x8  __attribute__((ext_vector_type(8)));
typedef __bf16 bf16x16 __attribute__((ext_vector_type(16)));
typedef float  f32x8   __attribute__((ext_vector_type(8)));

__device__ __forceinline__ u16 f2bf(float f) {
  unsigned u = __float_as_uint(f);
  unsigned r = u + 0x7FFFu + ((u >> 16) & 1u);   // round-to-nearest-even
  return (u16)(r >> 16);
}

__device__ __forceinline__ bf16x16 frag_cat(uint4 a, uint4 b) {
  bf16x8 lo = __builtin_bit_cast(bf16x8, a);
  bf16x8 hi = __builtin_bit_cast(bf16x8, b);
  return __builtin_shufflevector(lo, hi, 0,1,2,3,4,5,6,7,8,9,10,11,12,13,14,15);
}

// ---------------- Setup kernels ----------------
__global__ void k_zero_out(float4* __restrict__ p, long n4) {
  long i = (long)blockIdx.x * blockDim.x + threadIdx.x;
  long stride = (long)gridDim.x * blockDim.x;
  float4 z; z.x = z.y = z.z = z.w = 0.f;
  for (; i < n4; i += stride) p[i] = z;
}

__global__ void k_init_small(int* __restrict__ p) {
  if (threadIdx.x < 32) p[threadIdx.x] = 0;   // counts[8], cnt2[8], base[8], pad
}

__global__ void k_cvt_x(const float* __restrict__ in, u16* __restrict__ out, long n) {
  long i = ((long)blockIdx.x * blockDim.x + threadIdx.x) * 4;
  if (i >= n) return;
  float4 v = *(const float4*)(in + i);
  ushort4 o; o.x = f2bf(v.x); o.y = f2bf(v.y); o.z = f2bf(v.z); o.w = f2bf(v.w);
  *(ushort4*)(out + i) = o;
}

// in: [E][R][C] fp32 row-major -> out: [E][C][R] bf16
__global__ void k_transpose_cvt(const float* __restrict__ in, u16* __restrict__ out,
                                int R, int C) {
  __shared__ float t[32][33];
  int e = blockIdx.z;
  const float* src = in + (size_t)e * R * C;
  u16* dst = out + (size_t)e * R * C;
  int c0 = blockIdx.x * 32, r0 = blockIdx.y * 32;
  int tx = threadIdx.x, ty = threadIdx.y;
  #pragma unroll
  for (int j = ty; j < 32; j += 8)
    t[j][tx] = src[(size_t)(r0 + j) * C + c0 + tx];
  __syncthreads();
  #pragma unroll
  for (int j = ty; j < 32; j += 8)
    dst[(size_t)(c0 + j) * R + r0 + tx] = f2bf(t[tx][j]);
}

// ---------------- Gating: logits, top-2, softmax ----------------
__global__ void k_gate(const float* __restrict__ x, const float* __restrict__ gw,
                       const float* __restrict__ gb, int* __restrict__ topIdx,
                       float* __restrict__ topGate, int* __restrict__ counts) {
  int lane = threadIdx.x & 31;
  int wv   = threadIdx.x >> 5;
  int n = blockIdx.x * 8 + wv;      // 8 waves/block, one token each
  float acc[NEXP];
  #pragma unroll
  for (int e = 0; e < NEXP; ++e) acc[e] = 0.f;
  const float* xr = x + (size_t)n * DDIM;
  for (int d0 = 0; d0 < DDIM; d0 += 32) {
    float xv = xr[d0 + lane];
    const float* g = gw + (size_t)(d0 + lane) * NEXP;
    #pragma unroll
    for (int e = 0; e < NEXP; ++e) acc[e] += xv * g[e];
  }
  __shared__ float red[8][NEXP][33];
  #pragma unroll
  for (int e = 0; e < NEXP; ++e) red[wv][e][lane] = acc[e];
  __syncthreads();
  if (lane == 0) {
    float lg[NEXP];
    #pragma unroll
    for (int e = 0; e < NEXP; ++e) {
      float s = 0.f;
      for (int l = 0; l < 32; ++l) s += red[wv][e][l];
      lg[e] = s + gb[e];
    }
    int e0 = 0; float v0 = lg[0];
    #pragma unroll
    for (int e = 1; e < NEXP; ++e) if (lg[e] > v0) { v0 = lg[e]; e0 = e; }
    int e1 = -1; float v1 = -3.4e38f;
    #pragma unroll
    for (int e = 0; e < NEXP; ++e) if (e != e0 && lg[e] > v1) { v1 = lg[e]; e1 = e; }
    float ex  = expf(v1 - v0);
    float inv = 1.f / (1.f + ex);
    topIdx[n * 2 + 0] = e0;  topIdx[n * 2 + 1] = e1;
    topGate[n * 2 + 0] = inv; topGate[n * 2 + 1] = ex * inv;
    atomicAdd(&counts[e0], 1);
    atomicAdd(&counts[e1], 1);
  }
}

__global__ void k_prefix(const int* __restrict__ counts, int* __restrict__ base) {
  if (threadIdx.x == 0) {
    int b = 0;
    for (int e = 0; e < NEXP; ++e) { base[e] = b; b += counts[e]; }
  }
}

__global__ void k_scatter(const int* __restrict__ topIdx, const float* __restrict__ topGate,
                          const int* __restrict__ base, int* __restrict__ cnt2,
                          int* __restrict__ tokIdx, float* __restrict__ gateVal) {
  int n = blockIdx.x * blockDim.x + threadIdx.x;
  if (n >= N_TOK) return;
  #pragma unroll
  for (int k = 0; k < 2; ++k) {
    int e = topIdx[n * 2 + k];
    int pos = atomicAdd(&cnt2[e], 1);
    int slot = base[e] + pos;
    tokIdx[slot] = n;
    gateVal[slot] = topGate[n * 2 + k];
  }
}

// --------- shared GEMM helpers (double-buffered bf16 WMMA pipeline) ---------
// Invalid staging rows are CLAMPED to the last valid row (always-valid global
// addresses -> pure global_load_b128, no flat/cndmask in the hot loop). The
// garbage rows only pollute their own C rows and are masked in the epilogue.
#define GEMM_PROLOG()                                                          \
  int tid  = threadIdx.x;                                                      \
  int lane = tid & 31, wv = tid >> 5;                                          \
  int wm = (wv & 1) * 64;                                                      \
  int wn = (wv >> 1) * 32;                                                     \
  int hiHalf = lane >> 4;                                                      \
  int l16    = lane & 15;                                                      \
  int ar0 = tid >> 2;                                                          \
  int ar1 = ar0 + 64;                                                          \
  int ac  = (tid & 3) * 8;                                                     \
  int cr0 = min(ar0, vr - 1);                                                  \
  int cr1 = min(ar1, vr - 1);                                                  \
  uint4 ra0, ra1, rb0, rb1;                                                    \
  f32x8 acc[4][2];                                                             \
  _Pragma("unroll")                                                            \
  for (int i = 0; i < 4; ++i)                                                  \
    _Pragma("unroll")                                                          \
    for (int j = 0; j < 2; ++j) acc[i][j] = (f32x8)0.f;

#define SSTORE(buf)                                                            \
  do {                                                                         \
    *(uint4*)&At[buf][ar0 * LDT + ac] = ra0;                                   \
    *(uint4*)&At[buf][ar1 * LDT + ac] = ra1;                                   \
    *(uint4*)&Bt[buf][ar0 * LDT + ac] = rb0;                                   \
    *(uint4*)&Bt[buf][ar1 * LDT + ac] = rb1;                                   \
  } while (0)

#define GEMM_COMPUTE(cur)                                                      \
  do {                                                                         \
    bf16x16 afr[4], bfr[2];                                                    \
    _Pragma("unroll")                                                          \
    for (int i = 0; i < 4; ++i) {                                              \
      const u16* p = &At[cur][(wm + i * 16 + l16) * LDT + hiHalf * 8];         \
      afr[i] = frag_cat(*(const uint4*)p, *(const uint4*)(p + 16));            \
    }                                                                          \
    _Pragma("unroll")                                                          \
    for (int j = 0; j < 2; ++j) {                                              \
      const u16* p = &Bt[cur][(wn + j * 16 + l16) * LDT + hiHalf * 8];         \
      bfr[j] = frag_cat(*(const uint4*)p, *(const uint4*)(p + 16));            \
    }                                                                          \
    _Pragma("unroll")                                                          \
    for (int i = 0; i < 4; ++i)                                                \
      _Pragma("unroll")                                                        \
      for (int j = 0; j < 2; ++j)                                              \
        acc[i][j] = __builtin_amdgcn_wmma_f32_16x16x32_bf16(                   \
            false, afr[i], false, bfr[j], (short)0, acc[i][j], false, false);  \
  } while (0)

// ---------------- GEMM1: h = relu(x @ w1^T + b1), grouped by expert ----------------
// xb:[N,D] bf16; w1t:[E,H,D] bf16; hb:[NSLOT,H] bf16
__global__ __launch_bounds__(256) void k_moe_gemm1(
    const u16* __restrict__ xb, const u16* __restrict__ w1t,
    const float* __restrict__ b1, const int* __restrict__ counts,
    const int* __restrict__ base, const int* __restrict__ tokIdx,
    u16* __restrict__ hb) {
  int e   = blockIdx.z;
  int cnt = counts[e];
  int t0  = blockIdx.y * TM;
  if (t0 >= cnt) return;
  int vr    = min(TM, cnt - t0);
  int slot0 = base[e] + t0;
  int hBase = blockIdx.x * TN;

  __shared__ u16 At[2][TM * LDT];
  __shared__ u16 Bt[2][TN * LDT];

  GEMM_PROLOG();

  const u16* asrc0 = xb + (size_t)tokIdx[slot0 + cr0] * DDIM;
  const u16* asrc1 = xb + (size_t)tokIdx[slot0 + cr1] * DDIM;
  const u16* bsrc  = w1t + ((size_t)e * HDIM + hBase) * DDIM;

#define GLOAD1(kt)                                                             \
  do {                                                                         \
    ra0 = *(const uint4*)(asrc0 + (kt) * KC + ac);                             \
    ra1 = *(const uint4*)(asrc1 + (kt) * KC + ac);                             \
    rb0 = *(const uint4*)(bsrc + (size_t)ar0 * DDIM + (kt) * KC + ac);         \
    rb1 = *(const uint4*)(bsrc + (size_t)ar1 * DDIM + (kt) * KC + ac);         \
  } while (0)

  const int NK = DDIM / KC;  // 32
  GLOAD1(0);
  SSTORE(0);
  GLOAD1(1);
  for (int kt = 0; kt < NK; ++kt) {
    __syncthreads();
    if (kt + 1 < NK) SSTORE((kt + 1) & 1);
    if (kt + 2 < NK) GLOAD1(kt + 2);
    GEMM_COMPUTE(kt & 1);
  }
#undef GLOAD1

  // epilogue: bias + relu + bf16 store
  if (vr == TM) {
    #pragma unroll
    for (int j = 0; j < 2; ++j) {
      int hcol = hBase + wn + j * 16 + l16;
      float bias = b1[(size_t)e * HDIM + hcol];
      #pragma unroll
      for (int i = 0; i < 4; ++i) {
        #pragma unroll
        for (int r = 0; r < 8; ++r) {
          int row = wm + i * 16 + r + hiHalf * 8;
          float v = acc[i][j][r] + bias;
          v = v > 0.f ? v : 0.f;
          hb[(size_t)(slot0 + row) * HDIM + hcol] = f2bf(v);
        }
      }
    }
  } else {
    #pragma unroll
    for (int j = 0; j < 2; ++j) {
      int hcol = hBase + wn + j * 16 + l16;
      float bias = b1[(size_t)e * HDIM + hcol];
      #pragma unroll
      for (int i = 0; i < 4; ++i) {
        #pragma unroll
        for (int r = 0; r < 8; ++r) {
          int row = wm + i * 16 + r + hiHalf * 8;
          if (row < vr) {
            float v = acc[i][j][r] + bias;
            v = v > 0.f ? v : 0.f;
            hb[(size_t)(slot0 + row) * HDIM + hcol] = f2bf(v);
          }
        }
      }
    }
  }
}

// ---------------- GEMM2: out[tok] += g * (h @ w2^T + b2) ----------------
// hb:[NSLOT,H] bf16; w2t:[E,O,H] bf16; out:[N,O] f32
__global__ __launch_bounds__(256) void k_moe_gemm2(
    const u16* __restrict__ hb, const u16* __restrict__ w2t,
    const float* __restrict__ b2, const int* __restrict__ counts,
    const int* __restrict__ base, const int* __restrict__ tokIdx,
    const float* __restrict__ gateVal, float* __restrict__ out) {
  int e   = blockIdx.z;
  int cnt = counts[e];
  int t0  = blockIdx.y * TM;
  if (t0 >= cnt) return;
  int vr    = min(TM, cnt - t0);
  int slot0 = base[e] + t0;
  int oBase = blockIdx.x * TN;

  __shared__ u16 At[2][TM * LDT];
  __shared__ u16 Bt[2][TN * LDT];

  GEMM_PROLOG();

  const u16* asrc0 = hb + (size_t)(slot0 + cr0) * HDIM;
  const u16* asrc1 = hb + (size_t)(slot0 + cr1) * HDIM;
  const u16* bsrc  = w2t + ((size_t)e * ODIM + oBase) * HDIM;

#define GLOAD2(kt)                                                             \
  do {                                                                         \
    ra0 = *(const uint4*)(asrc0 + (kt) * KC + ac);                             \
    ra1 = *(const uint4*)(asrc1 + (kt) * KC + ac);                             \
    rb0 = *(const uint4*)(bsrc + (size_t)ar0 * HDIM + (kt) * KC + ac);         \
    rb1 = *(const uint4*)(bsrc + (size_t)ar1 * HDIM + (kt) * KC + ac);         \
  } while (0)

  const int NK = HDIM / KC;  // 64
  GLOAD2(0);
  SSTORE(0);
  GLOAD2(1);
  for (int kt = 0; kt < NK; ++kt) {
    __syncthreads();
    if (kt + 1 < NK) SSTORE((kt + 1) & 1);
    if (kt + 2 < NK) GLOAD2(kt + 2);
    GEMM_COMPUTE(kt & 1);
  }
#undef GLOAD2

  float bias[2]; int ocol[2];
  #pragma unroll
  for (int j = 0; j < 2; ++j) {
    ocol[j] = oBase + wn + j * 16 + l16;
    bias[j] = b2[(size_t)e * ODIM + ocol[j]];
  }
  if (vr == TM) {
    #pragma unroll
    for (int i = 0; i < 4; ++i) {
      #pragma unroll
      for (int r = 0; r < 8; ++r) {
        int row = wm + i * 16 + r + hiHalf * 8;
        int   tok = tokIdx[slot0 + row];
        float g   = gateVal[slot0 + row];
        #pragma unroll
        for (int j = 0; j < 2; ++j)
          atomicAdd(&out[(size_t)tok * ODIM + ocol[j]], g * (acc[i][j][r] + bias[j]));
      }
    }
  } else {
    #pragma unroll
    for (int i = 0; i < 4; ++i) {
      #pragma unroll
      for (int r = 0; r < 8; ++r) {
        int row = wm + i * 16 + r + hiHalf * 8;
        if (row < vr) {
          int   tok = tokIdx[slot0 + row];
          float g   = gateVal[slot0 + row];
          #pragma unroll
          for (int j = 0; j < 2; ++j)
            atomicAdd(&out[(size_t)tok * ODIM + ocol[j]], g * (acc[i][j][r] + bias[j]));
        }
      }
    }
  }
}

// ---------------- Host launcher ----------------
extern "C" void kernel_launch(void* const* d_in, const int* in_sizes, int n_in,
                              void* d_out, int out_size, void* d_ws, size_t ws_size,
                              hipStream_t stream) {
  const float* x      = (const float*)d_in[0];
  const float* gate_w = (const float*)d_in[1];
  const float* gate_b = (const float*)d_in[2];
  const float* w1     = (const float*)d_in[3];
  const float* b1     = (const float*)d_in[4];
  const float* w2     = (const float*)d_in[5];
  const float* b2     = (const float*)d_in[6];
  float* out = (float*)d_out;

  // workspace carve-out (~145 MB)
  char* ws = (char*)d_ws;
  u16* xb  = (u16*)ws; ws += (size_t)N_TOK * DDIM * 2;
  u16* w1t = (u16*)ws; ws += (size_t)NEXP * DDIM * HDIM * 2;
  u16* w2t = (u16*)ws; ws += (size_t)NEXP * HDIM * ODIM * 2;
  u16* hb  = (u16*)ws; ws += (size_t)NSLOT * HDIM * 2;
  int*   topIdx  = (int*)ws;   ws += (size_t)N_TOK * 2 * sizeof(int);
  float* topGate = (float*)ws; ws += (size_t)N_TOK * 2 * sizeof(float);
  int*   tokIdx  = (int*)ws;   ws += (size_t)NSLOT * sizeof(int);
  float* gateVal = (float*)ws; ws += (size_t)NSLOT * sizeof(float);
  int* smallInts = (int*)ws;               // counts[8] | cnt2[8] | base[8]
  int* counts = smallInts;
  int* cnt2   = smallInts + 8;
  int* base   = smallInts + 16;

  // 1) zero output + routing counters (every call: harness does not re-poison)
  long n4 = (long)out_size / 4;
  k_zero_out<<<2048, 256, 0, stream>>>((float4*)out, n4);
  k_init_small<<<1, 64, 0, stream>>>(smallInts);

  // 2) precision/layout conversion
  long nx = (long)N_TOK * DDIM;
  k_cvt_x<<<(int)(nx / 4 / 256), 256, 0, stream>>>(x, xb, nx);
  k_transpose_cvt<<<dim3(HDIM / 32, DDIM / 32, NEXP), dim3(32, 8), 0, stream>>>(w1, w1t, DDIM, HDIM);
  k_transpose_cvt<<<dim3(ODIM / 32, HDIM / 32, NEXP), dim3(32, 8), 0, stream>>>(w2, w2t, HDIM, ODIM);

  // 3) gating + grouped dispatch
  k_gate<<<N_TOK / 8, 256, 0, stream>>>(x, gate_w, gate_b, topIdx, topGate, counts);
  k_prefix<<<1, 32, 0, stream>>>(counts, base);
  k_scatter<<<N_TOK / 256, 256, 0, stream>>>(topIdx, topGate, base, cnt2, tokIdx, gateVal);

  // 4) grouped expert GEMMs (y-dim covers worst case: one expert gets all tokens)
  k_moe_gemm1<<<dim3(HDIM / TN, N_TOK / TM, NEXP), 256, 0, stream>>>(
      xb, w1t, b1, counts, base, tokIdx, hb);
  k_moe_gemm2<<<dim3(ODIM / TN, N_TOK / TM, NEXP), 256, 0, stream>>>(
      hb, w2t, b2, counts, base, tokIdx, gateVal, out);
}
